// MMoEMatcher_20323785244720
// MI455X (gfx1250) — matrix operand order repaired and checked by reading, not compiled, not run on hardware
//
#include <hip/hip_runtime.h>
#include <stdint.h>

// ---------------- types / helpers ----------------
typedef __attribute__((ext_vector_type(16))) __bf16 v16bf;
typedef __attribute__((ext_vector_type(2)))  __bf16 v2bf;
typedef __attribute__((ext_vector_type(8)))  float  v8f;
typedef __attribute__((ext_vector_type(2)))  float  v2f;

union FragBF { uint32_t u[8]; v16bf v; };

// fp32 -> bf16 via clang native conversion (lets the backend pick the
// hardware cvt op on gfx1250 instead of a manual integer RNE sequence)
__device__ __forceinline__ uint16_t f2bf(float f) {
    union { __bf16 b; uint16_t u; } c; c.b = (__bf16)f; return c.u;
}
__device__ __forceinline__ float bf2f(uint16_t h) {
    union { uint32_t u; float f; } c; c.u = ((uint32_t)h) << 16;
    return c.f;
}
__device__ __forceinline__ uint32_t pack2bf(float a, float b) {
#if defined(__gfx1250__) && __has_builtin(__builtin_amdgcn_cvt_pk_bf16_f32)
    union { v2bf v; uint32_t u; } c;
    c.v = __builtin_amdgcn_cvt_pk_bf16_f32(a, b);
    return c.u;
#else
    union { v2bf v; uint32_t u; } c;
    c.v[0] = (__bf16)a; c.v[1] = (__bf16)b;
    return c.u;
#endif
}

// wait for outstanding async (global->LDS) transfers of this wave
__device__ __forceinline__ void async_wait0() {
#if defined(__gfx1250__)
    asm volatile("s_wait_asynccnt 0" ::: "memory");
#endif
}

#define TM   128   // block tile M
#define TN   64    // block tile N
#define TK   32    // k-step (== WMMA bf16 K)
#define LSK  40    // padded LDS k-stride: 80B rows (16B aligned, 20-bank skew)
#define HDIM 512

// ============================================================
// Weight pre-transpose: in [B][K][N] fp32 -> out [B][N][K] bf16
// ============================================================
__global__ __launch_bounds__(256) void wtrans_bf16_kernel(
    const float* __restrict__ in, uint16_t* __restrict__ out, int K, int N)
{
    const int e = blockIdx.y;
    const float* src = in + (size_t)e * K * N;
    uint16_t*   dst = out + (size_t)e * N * K;
    const int halfK = K >> 1;
    int idx = blockIdx.x * 256 + threadIdx.x;       // one bf16-pair per thread
    if (idx < N * halfK) {
        int n = idx / halfK;
        int k = (idx - n * halfK) << 1;
        uint32_t p = pack2bf(src[(size_t)k * N + n], src[(size_t)(k + 1) * N + n]);
        *(uint32_t*)&dst[(size_t)n * K + k] = p;
    }
}

// ============================================================
// Projection GEMM: Y = X(fp32, MxK) @ W(bf16 pre-transposed [H][K]) + bias
// bf16 out scattered into concat [cls ; tokens] layout; optional fp32
// out (flat [M][512]) kept only for cls rows.
// ============================================================
__global__ __launch_bounds__(256) void proj_gemm_kernel(
    const float* __restrict__ X, const uint16_t* __restrict__ WbfT,
    const float* __restrict__ bias,
    float* __restrict__ Yf,              // nullptr for token streams
    uint16_t* __restrict__ Ybf,
    int M, int K, int S_tok, int S_full, int row_off)
{
    __shared__ uint16_t Alds[2][TM][LSK];
    __shared__ uint16_t Blds[2][TN][LSK];

    const int tid   = threadIdx.x;
    const int lane  = tid & 31;
    const int wave  = tid >> 5;
    const int blkm  = blockIdx.x * TM;
    const int n0    = blockIdx.y * TN;
    const int nrow  = lane & 15;
    const int khalf = lane >> 4;

    v8f acc[4];
    #pragma unroll
    for (int t = 0; t < 4; ++t)
        #pragma unroll
        for (int j = 0; j < 8; ++j) acc[t][j] = 0.0f;

    auto loadA = [&](int buf, int k0) {
        #pragma unroll
        for (int j = 0; j < 4; ++j) {
            int idx = tid + 256 * j;
            int row = idx >> 3;
            int c4  = (idx & 7) << 2;
            int r   = blkm + row;
            float4 d = make_float4(0.f, 0.f, 0.f, 0.f);
            if (r < M) {
                const float* p = X + (size_t)r * K + k0 + c4;
                d = *(const float4*)p;
                if (k0 + 2 * TK < K) __builtin_prefetch(p + 2 * TK, 0, 1);
            }
            *(uint32_t*)&Alds[buf][row][c4]     = pack2bf(d.x, d.y);
            *(uint32_t*)&Alds[buf][row][c4 + 2] = pack2bf(d.z, d.w);
        }
    };
    // B tile is a verbatim bf16 copy: use the CDNA5 async global->LDS path.
    auto loadB = [&](int buf, int k0) {
#if defined(__gfx1250__)
        int n  = tid >> 2;             // 64 rows, 4 x 16B chunks per row
        int c8 = (tid & 3) << 3;       // bf16 elems: 0,8,16,24
        uint32_t lds = (uint32_t)(uintptr_t)&Blds[buf][n][c8];
        const uint16_t* p = WbfT + (size_t)(n0 + n) * K + k0 + c8;
        asm volatile("global_load_async_to_lds_b128 %0, %1, off"
                     :: "v"(lds), "v"((uint64_t)(uintptr_t)p) : "memory");
#else
        #pragma unroll
        for (int j = 0; j < 2; ++j) {
            int idx = tid + 256 * j;
            int n   = idx >> 3;
            int c4  = (idx & 7) << 2;
            const uint16_t* p = WbfT + (size_t)(n0 + n) * K + k0 + c4;
            *(uint2*)&Blds[buf][n][c4] = *(const uint2*)p;
        }
#endif
    };
    auto compute = [&](int buf) {
        FragBF a;
        #pragma unroll
        for (int v = 0; v < 8; ++v) {
            int kk = ((v & 4) << 2) + khalf * 8 + ((v & 3) << 1);
            a.u[v] = *(const uint32_t*)&Alds[buf][wave * 16 + nrow][kk];
        }
        #pragma unroll
        for (int nt = 0; nt < 4; ++nt) {
            FragBF b;
            #pragma unroll
            for (int v = 0; v < 8; ++v) {
                int kk = ((v & 4) << 2) + khalf * 8 + ((v & 3) << 1);
                b.u[v] = *(const uint32_t*)&Blds[buf][nt * 16 + nrow][kk];
            }
            acc[nt] = __builtin_amdgcn_wmma_f32_16x16x32_bf16(
                false, a.v, false, b.v, (short)0, acc[nt], false, false);
        }
    };

    const int nk = K / TK;
    loadA(0, 0); loadB(0, 0);
    async_wait0();
    __syncthreads();
    for (int kt = 0; kt < nk; ++kt) {
        const int cur = kt & 1;
        if (kt + 1 < nk) { loadA(cur ^ 1, (kt + 1) * TK); loadB(cur ^ 1, (kt + 1) * TK); }
        compute(cur);
        async_wait0();
        __syncthreads();
    }

    #pragma unroll
    for (int nt = 0; nt < 4; ++nt) {
        #pragma unroll
        for (int i = 0; i < 8; ++i) {
            int r = blkm + wave * 16 + i + khalf * 8;
            if (r < M) {
                int col = n0 + nt * 16 + nrow;
                int bb  = r / S_tok;
                int tt  = r - bb * S_tok;
                size_t orow = (size_t)bb * S_full + row_off + tt;
                float val = acc[nt][i] + bias[col];
                Ybf[orow * HDIM + col] = f2bf(val);
                if (Yf) Yf[(size_t)r * HDIM + col] = val;
            }
        }
    }
}

// ============================================================
// Routing: mean over S tokens (bf16 stream) -> logits -> softmax ->
// top-2 -> normalized mask[nb][8] + combined bias[nb][512]
// ============================================================
__global__ __launch_bounds__(256) void routing_kernel(
    const uint16_t* __restrict__ Y,   // [nb][S][512] bf16
    const float* __restrict__ Wr,     // [512][8]
    const float* __restrict__ br,     // [8]
    const float* __restrict__ be,     // [8][512]
    float* __restrict__ mask,         // [nb][8]
    float* __restrict__ bias_comb,    // [nb][512]
    int S)
{
    __shared__ float meanv[HDIM];
    __shared__ float logits[8];
    __shared__ float lmask[8];
    const int b   = blockIdx.x;
    const int tid = threadIdx.x;
    const uint16_t* yb = Y + (size_t)b * S * HDIM;

    {
        int c = tid * 2;                  // 256 threads x 2 cols
        float s0 = 0.f, s1 = 0.f;
        for (int t = 0; t < S; ++t) {
            uint32_t d = *(const uint32_t*)(yb + (size_t)t * HDIM + c);
            s0 += bf2f((uint16_t)(d & 0xffffu));
            s1 += bf2f((uint16_t)(d >> 16));
        }
        meanv[c]     = s0 / (float)S;
        meanv[c + 1] = s1 / (float)S;
    }
    __syncthreads();
    if (tid < 8) {
        float s = br[tid];
        for (int d = 0; d < HDIM; ++d) s += meanv[d] * Wr[d * 8 + tid];
        logits[tid] = s;
    }
    __syncthreads();
    if (tid == 0) {
        float mx = logits[0];
        for (int e = 1; e < 8; ++e) mx = fmaxf(mx, logits[e]);
        float pr[8]; float sum = 0.f;
        for (int e = 0; e < 8; ++e) { pr[e] = __expf(logits[e] - mx); sum += pr[e]; }
        for (int e = 0; e < 8; ++e) pr[e] /= sum;
        int i0 = 0;
        for (int e = 1; e < 8; ++e) if (pr[e] > pr[i0]) i0 = e;
        int i1 = (i0 == 0) ? 1 : 0;
        for (int e = 0; e < 8; ++e) if (e != i0 && pr[e] > pr[i1]) i1 = e;
        float ws = pr[i0] + pr[i1];
        for (int e = 0; e < 8; ++e) lmask[e] = 0.f;
        lmask[i0] = pr[i0] / ws;
        lmask[i1] = pr[i1] / ws;
        for (int e = 0; e < 8; ++e) mask[(size_t)b * 8 + e] = lmask[e];
    }
    __syncthreads();
    for (int c = tid; c < HDIM; c += 256) {
        float s = 0.f;
        for (int e = 0; e < 8; ++e) s += lmask[e] * be[e * HDIM + c];
        bias_comb[(size_t)b * HDIM + c] = s;
    }
}

// ============================================================
// MoE GEMM: Out = sum_e diag(mask[:,e]) * (X @ We[e]) + bias_comb
// Expert weights pre-transposed bf16 [8][N][K]; rows scaled by mask
// while staged into LDS; zero-mask tiles skip the expert entirely.
// ============================================================
__global__ __launch_bounds__(256) void moe_gemm_kernel(
    const uint16_t* __restrict__ Abf,     // [M][512] bf16
    const uint16_t* __restrict__ WexpT,   // [8][512][512] bf16, n-major
    const float* __restrict__ mask,       // [nb][8]
    const float* __restrict__ bias_comb,  // [nb][512]
    float* __restrict__ Out,              // [M][512] fp32
    int M, int S)
{
    __shared__ uint16_t Alds[2][TM][LSK];
    __shared__ uint16_t Blds[2][TN][LSK];
    __shared__ float    rscale[TM];
    __shared__ int      eflag;

    const int tid   = threadIdx.x;
    const int lane  = tid & 31;
    const int wave  = tid >> 5;
    const int blkm  = blockIdx.x * TM;
    const int n0    = blockIdx.y * TN;
    const int nrow  = lane & 15;
    const int khalf = lane >> 4;

    v8f acc[4];
    #pragma unroll
    for (int t = 0; t < 4; ++t)
        #pragma unroll
        for (int j = 0; j < 8; ++j) acc[t][j] = 0.0f;

    auto loadA = [&](int buf, int k0) {
        #pragma unroll
        for (int j = 0; j < 4; ++j) {
            int idx = tid + 256 * j;
            int row = idx >> 3;
            int c4  = (idx & 7) << 2;
            int r   = blkm + row;
            uint2 d = make_uint2(0u, 0u);
            if (r < M) {
                const uint16_t* p = Abf + (size_t)r * HDIM + k0 + c4;
                d = *(const uint2*)p;
                if (k0 + 2 * TK < HDIM) __builtin_prefetch(p + 2 * TK, 0, 1);
            }
            float s = rscale[row];
            *(uint32_t*)&Alds[buf][row][c4] =
                pack2bf(bf2f((uint16_t)(d.x & 0xffffu)) * s,
                        bf2f((uint16_t)(d.x >> 16)) * s);
            *(uint32_t*)&Alds[buf][row][c4 + 2] =
                pack2bf(bf2f((uint16_t)(d.y & 0xffffu)) * s,
                        bf2f((uint16_t)(d.y >> 16)) * s);
        }
    };
    auto loadB = [&](int buf, int k0, int e) {
#if defined(__gfx1250__)
        int n  = tid >> 2;
        int c8 = (tid & 3) << 3;
        uint32_t lds = (uint32_t)(uintptr_t)&Blds[buf][n][c8];
        const uint16_t* p = WexpT + ((size_t)e * HDIM + (n0 + n)) * HDIM + k0 + c8;
        asm volatile("global_load_async_to_lds_b128 %0, %1, off"
                     :: "v"(lds), "v"((uint64_t)(uintptr_t)p) : "memory");
#else
        #pragma unroll
        for (int j = 0; j < 2; ++j) {
            int idx = tid + 256 * j;
            int n   = idx >> 3;
            int c4  = (idx & 7) << 2;
            const uint16_t* p = WexpT + ((size_t)e * HDIM + (n0 + n)) * HDIM + k0 + c4;
            *(uint2*)&Blds[buf][n][c4] = *(const uint2*)p;
        }
#endif
    };
    auto compute = [&](int buf) {
        FragBF a;
        #pragma unroll
        for (int v = 0; v < 8; ++v) {
            int kk = ((v & 4) << 2) + khalf * 8 + ((v & 3) << 1);
            a.u[v] = *(const uint32_t*)&Alds[buf][wave * 16 + nrow][kk];
        }
        #pragma unroll
        for (int nt = 0; nt < 4; ++nt) {
            FragBF b;
            #pragma unroll
            for (int v = 0; v < 8; ++v) {
                int kk = ((v & 4) << 2) + khalf * 8 + ((v & 3) << 1);
                b.u[v] = *(const uint32_t*)&Blds[buf][nt * 16 + nrow][kk];
            }
            acc[nt] = __builtin_amdgcn_wmma_f32_16x16x32_bf16(
                false, a.v, false, b.v, (short)0, acc[nt], false, false);
        }
    };

    for (int e = 0; e < 8; ++e) {
        if (tid == 0) eflag = 0;
        __syncthreads();
        if (tid < TM) {
            int r = blkm + tid;
            float sc = 0.f;
            if (r < M) sc = mask[(size_t)(r / S) * 8 + e];
            rscale[tid] = sc;
            if (sc != 0.f) eflag = 1;   // benign race: all writers store 1
        }
        __syncthreads();
        if (eflag == 0) continue;        // uniform: whole tile skips expert

        const int nk = HDIM / TK;        // 16
        loadA(0, 0); loadB(0, 0, e);
        async_wait0();
        __syncthreads();
        for (int kt = 0; kt < nk; ++kt) {
            const int cur = kt & 1;
            if (kt + 1 < nk) { loadA(cur ^ 1, (kt + 1) * TK); loadB(cur ^ 1, (kt + 1) * TK, e); }
            compute(cur);
            async_wait0();
            __syncthreads();
        }
    }

    #pragma unroll
    for (int nt = 0; nt < 4; ++nt) {
        #pragma unroll
        for (int i = 0; i < 8; ++i) {
            int r = blkm + wave * 16 + i + khalf * 8;
            if (r < M) {
                int col = n0 + nt * 16 + nrow;
                Out[(size_t)r * HDIM + col] =
                    acc[nt][i] + bias_comb[(size_t)(r / S) * HDIM + col];
            }
        }
    }
}

// ============================================================
// Cross-attn + tanh gate + LayerNorm per row.
// gate_on_ctx=1 -> gate(ctx) (entities); 0 -> gate(cls_ori) (mentions)
// ============================================================
__global__ __launch_bounds__(256) void cross_gate_ln_kernel(
    const float* __restrict__ Moe,     // [nb][S][512] moe output
    const float* __restrict__ OriCls,  // [nb][512] fp32 original cls
    const float* __restrict__ Wg,      // [512]
    const float* __restrict__ bg,      // [1]
    const float* __restrict__ lns,     // [512]
    const float* __restrict__ lnb,     // [512]
    float* __restrict__ OutCtx,        // [nb][512]
    int S, int gate_on_ctx)
{
    __shared__ float cls2[HDIM];
    __shared__ float ctx[HDIM];
    __shared__ float sc[64];
    __shared__ float red[256];
    __shared__ float sg;
    const int b   = blockIdx.x;
    const int tid = threadIdx.x;
    const float* mb  = Moe + (size_t)b * S * HDIM;
    const float* tok = mb + HDIM;
    const float* ori = OriCls + (size_t)b * HDIM;
    const int ns = S - 1;

    for (int c = tid; c < HDIM; c += 256) cls2[c] = mb[c];
    __syncthreads();
    if (tid < ns) {
        float s = 0.f;
        const float* tp = tok + (size_t)tid * HDIM;
        for (int d = 0; d < HDIM; ++d) s += cls2[d] * tp[d];
        sc[tid] = s;
    }
    __syncthreads();
    if (tid == 0) {
        float mx = sc[0];
        for (int t = 1; t < ns; ++t) mx = fmaxf(mx, sc[t]);
        float sum = 0.f;
        for (int t = 0; t < ns; ++t) { float p = __expf(sc[t] - mx); sc[t] = p; sum += p; }
        float inv = 1.f / sum;
        for (int t = 0; t < ns; ++t) sc[t] *= inv;
    }
    __syncthreads();
    for (int c = tid; c < HDIM; c += 256) {
        float s = 0.f;
        for (int t = 0; t < ns; ++t) s += sc[t] * tok[(size_t)t * HDIM + c];
        ctx[c] = s;
    }
    __syncthreads();
    float part = 0.f;
    for (int c = tid; c < HDIM; c += 256) {
        float gsrc = gate_on_ctx ? ctx[c] : ori[c];
        part += gsrc * Wg[c];
    }
    red[tid] = part; __syncthreads();
    for (int s2 = 128; s2 > 0; s2 >>= 1) { if (tid < s2) red[tid] += red[tid + s2]; __syncthreads(); }
    if (tid == 0) sg = tanhf(red[0] + bg[0]);
    __syncthreads();
    const float g = sg;
    __syncthreads();
    float pm = 0.f;
    for (int c = tid; c < HDIM; c += 256) { float y = ori[c] * g + ctx[c]; ctx[c] = y; pm += y; }
    red[tid] = pm; __syncthreads();
    for (int s2 = 128; s2 > 0; s2 >>= 1) { if (tid < s2) red[tid] += red[tid + s2]; __syncthreads(); }
    const float mean = red[0] / (float)HDIM;
    __syncthreads();
    float pv = 0.f;
    for (int c = tid; c < HDIM; c += 256) { float d = ctx[c] - mean; pv += d * d; }
    red[tid] = pv; __syncthreads();
    for (int s2 = 128; s2 > 0; s2 >>= 1) { if (tid < s2) red[tid] += red[tid + s2]; __syncthreads(); }
    const float rinv = rsqrtf(red[0] / (float)HDIM + 1e-5f);
    __syncthreads();
    for (int c = tid; c < HDIM; c += 256)
        OutCtx[(size_t)b * HDIM + c] = (ctx[c] - mean) * rinv * lns[c] + lnb[c];
}

// ============================================================
// Score: Out(128x1024) = 0.5*(Mti@Eti^T + Mit@Eit^T), fp32 WMMA
// ============================================================
#if defined(__has_builtin)
#  if __has_builtin(__builtin_amdgcn_wmma_f32_16x16x4_f32)
#    define HAVE_WMMA_F32X4 1
#  endif
#endif
#ifndef HAVE_WMMA_F32X4
#  define HAVE_WMMA_F32X4 0
#endif

__global__ __launch_bounds__(256) void score_kernel(
    const float* __restrict__ Mti, const float* __restrict__ Eti,
    const float* __restrict__ Mit, const float* __restrict__ Eit,
    float* __restrict__ Out)
{
    const int tid   = threadIdx.x;
    const int lane  = tid & 31;
    const int wave  = tid >> 5;
    const int wid   = blockIdx.x * 8 + wave;   // 512 wave-tiles
    const int tm    = (wid & 7) * 16;          // 128 / 16
    const int tn    = (wid >> 3) * 16;         // 1024 / 16
    const int nrow  = lane & 15;
    const int khalf = lane >> 4;
#if HAVE_WMMA_F32X4
    v8f acc;
    #pragma unroll
    for (int j = 0; j < 8; ++j) acc[j] = 0.f;
    const float* a_ti = Mti + (size_t)(tm + nrow) * HDIM + khalf * 2;
    const float* b_ti = Eti + (size_t)(tn + nrow) * HDIM + khalf * 2;
    const float* a_it = Mit + (size_t)(tm + nrow) * HDIM + khalf * 2;
    const float* b_it = Eit + (size_t)(tn + nrow) * HDIM + khalf * 2;
    for (int k0 = 0; k0 < HDIM; k0 += 4) {
        v2f a, b;
        a[0] = a_ti[k0]; a[1] = a_ti[k0 + 1];
        b[0] = b_ti[k0]; b[1] = b_ti[k0 + 1];
        acc = __builtin_amdgcn_wmma_f32_16x16x4_f32(false, a, false, b, (short)0, acc, false, false);
        a[0] = a_it[k0]; a[1] = a_it[k0 + 1];
        b[0] = b_it[k0]; b[1] = b_it[k0 + 1];
        acc = __builtin_amdgcn_wmma_f32_16x16x4_f32(false, a, false, b, (short)0, acc, false, false);
    }
    #pragma unroll
    for (int i = 0; i < 8; ++i) {
        int row = tm + i + khalf * 8;
        int col = tn + nrow;
        Out[(size_t)row * 1024 + col] = 0.5f * acc[i];
    }
#else
    for (int i = 0; i < 8; ++i) {
        int row = tm + i + khalf * 8;
        int col = tn + nrow;
        const float* a1 = Mti + (size_t)row * HDIM;
        const float* b1 = Eti + (size_t)col * HDIM;
        const float* a2 = Mit + (size_t)row * HDIM;
        const float* b2 = Eit + (size_t)col * HDIM;
        float s = 0.f;
        for (int k = 0; k < HDIM; ++k) s += a1[k] * b1[k] + a2[k] * b2[k];
        Out[(size_t)row * 1024 + col] = 0.5f * s;
    }
#endif
}

// ============================================================
// Host launcher
// ============================================================
extern "C" void kernel_launch(void* const* d_in, const int* in_sizes, int n_in,
                              void* d_out, int out_size, void* d_ws, size_t ws_size,
                              hipStream_t stream)
{
    (void)in_sizes; (void)n_in; (void)out_size; (void)ws_size;

    const float* x_etc = (const float*)d_in[0];
    const float* x_ett = (const float*)d_in[1];
    const float* x_mtc = (const float*)d_in[2];
    const float* x_mtt = (const float*)d_in[3];
    const float* x_eic = (const float*)d_in[4];
    const float* x_eit = (const float*)d_in[5];
    const float* x_mic = (const float*)d_in[6];
    const float* x_mit = (const float*)d_in[7];
    const float* Wt  = (const float*)d_in[8];
    const float* bt  = (const float*)d_in[9];
    const float* Wi  = (const float*)d_in[10];
    const float* bi  = (const float*)d_in[11];
    const float* Wg  = (const float*)d_in[12];
    const float* bg  = (const float*)d_in[13];
    const float* lns = (const float*)d_in[14];
    const float* lnb = (const float*)d_in[15];
    const float* Wr  = (const float*)d_in[16];
    const float* br  = (const float*)d_in[17];
    const float* We  = (const float*)d_in[18];
    const float* be  = (const float*)d_in[19];
    float* out = (float*)d_out;

    const int NE = 1024, NM = 128, STI = 50, SIT = 65, H = 512, DIN = 768;

    char* ws = (char*)d_ws;
    size_t off = 0;
    auto take = [&](size_t nbytes) -> void* {
        void* p = (void*)(ws + off);
        off += (nbytes + 255) & ~(size_t)255;
        return p;
    };

    uint16_t* e_ti_b = (uint16_t*)take((size_t)NE * STI * H * 2);
    uint16_t* e_it_b = (uint16_t*)take((size_t)NE * SIT * H * 2);
    uint16_t* m_ti_b = (uint16_t*)take((size_t)NM * STI * H * 2);
    uint16_t* m_it_b = (uint16_t*)take((size_t)NM * SIT * H * 2);
    float*    e_ti_m = (float*)   take((size_t)NE * STI * H * 4);
    float*    e_it_m = (float*)   take((size_t)NE * SIT * H * 4);
    float*    m_ti_m = (float*)   take((size_t)NM * STI * H * 4);
    float*    m_it_m = (float*)   take((size_t)NM * SIT * H * 4);
    float*    etc_ori = (float*)take((size_t)NE * H * 4);
    float*    eic_ori = (float*)take((size_t)NE * H * 4);
    float*    mtc_ori = (float*)take((size_t)NM * H * 4);
    float*    mic_ori = (float*)take((size_t)NM * H * 4);
    uint16_t* WtT = (uint16_t*)take((size_t)H * DIN * 2);
    uint16_t* WiT = (uint16_t*)take((size_t)H * DIN * 2);
    uint16_t* WeT = (uint16_t*)take((size_t)8 * H * H * 2);
    float* mask_eti = (float*)take((size_t)NE * 8 * 4);
    float* mask_mti = (float*)take((size_t)NM * 8 * 4);
    float* mask_eit = (float*)take((size_t)NE * 8 * 4);
    float* mask_mit = (float*)take((size_t)NM * 8 * 4);
    float* bias_eti = (float*)take((size_t)NE * H * 4);
    float* bias_mti = (float*)take((size_t)NM * H * 4);
    float* bias_eit = (float*)take((size_t)NE * H * 4);
    float* bias_mit = (float*)take((size_t)NM * H * 4);
    float* ctx_e_ti = (float*)take((size_t)NE * H * 4);
    float* ctx_m_ti = (float*)take((size_t)NM * H * 4);
    float* ctx_e_it = (float*)take((size_t)NE * H * 4);
    float* ctx_m_it = (float*)take((size_t)NM * H * 4);

    dim3 blk(256);
    auto pgrid = [](int M) { return dim3((M + TM - 1) / TM, H / TN); };

    // ---- Stage 0: weight pre-transpose (fp32 KxN -> bf16 NxK) ----
    wtrans_bf16_kernel<<<dim3((H * (DIN / 2) + 255) / 256, 1), blk, 0, stream>>>(Wt, WtT, DIN, H);
    wtrans_bf16_kernel<<<dim3((H * (DIN / 2) + 255) / 256, 1), blk, 0, stream>>>(Wi, WiT, DIN, H);
    wtrans_bf16_kernel<<<dim3((H * (H / 2) + 255) / 256, 8),   blk, 0, stream>>>(We, WeT, H, H);

    // ---- Stage 1: projections, scattered into concat layout ----
    proj_gemm_kernel<<<pgrid(NE),      blk, 0, stream>>>(x_etc, WtT, bt, etc_ori, e_ti_b, NE,      DIN, 1,  STI, 0);
    proj_gemm_kernel<<<pgrid(NE * 49), blk, 0, stream>>>(x_eit, WiT, bi, nullptr, e_ti_b, NE * 49, DIN, 49, STI, 1);
    proj_gemm_kernel<<<pgrid(NE),      blk, 0, stream>>>(x_eic, WiT, bi, eic_ori, e_it_b, NE,      DIN, 1,  SIT, 0);
    proj_gemm_kernel<<<pgrid(NE * 64), blk, 0, stream>>>(x_ett, WtT, bt, nullptr, e_it_b, NE * 64, DIN, 64, SIT, 1);
    proj_gemm_kernel<<<pgrid(NM),      blk, 0, stream>>>(x_mtc, WtT, bt, mtc_ori, m_ti_b, NM,      DIN, 1,  STI, 0);
    proj_gemm_kernel<<<pgrid(NM * 49), blk, 0, stream>>>(x_mit, WiT, bi, nullptr, m_ti_b, NM * 49, DIN, 49, STI, 1);
    proj_gemm_kernel<<<pgrid(NM),      blk, 0, stream>>>(x_mic, WiT, bi, mic_ori, m_it_b, NM,      DIN, 1,  SIT, 0);
    proj_gemm_kernel<<<pgrid(NM * 64), blk, 0, stream>>>(x_mtt, WtT, bt, nullptr, m_it_b, NM * 64, DIN, 64, SIT, 1);

    // ---- Stage 2: routing (mask + combined bias) ----
    routing_kernel<<<dim3(NE), blk, 0, stream>>>(e_it_b, Wr, br, be, mask_eti, bias_eti, SIT); // moe(e_ti, e_it)
    routing_kernel<<<dim3(NM), blk, 0, stream>>>(m_it_b, Wr, br, be, mask_mti, bias_mti, SIT); // moe(m_ti, m_it)
    routing_kernel<<<dim3(NE), blk, 0, stream>>>(e_ti_b, Wr, br, be, mask_eit, bias_eit, STI); // moe(e_it, e_ti)
    routing_kernel<<<dim3(NM), blk, 0, stream>>>(m_ti_b, Wr, br, be, mask_mit, bias_mit, STI); // moe(m_it, m_ti)

    // ---- Stage 3: MoE expert GEMMs ----
    moe_gemm_kernel<<<pgrid(NE * STI), blk, 0, stream>>>(e_ti_b, WeT, mask_eti, bias_eti, e_ti_m, NE * STI, STI);
    moe_gemm_kernel<<<pgrid(NM * STI), blk, 0, stream>>>(m_ti_b, WeT, mask_mti, bias_mti, m_ti_m, NM * STI, STI);
    moe_gemm_kernel<<<pgrid(NE * SIT), blk, 0, stream>>>(e_it_b, WeT, mask_eit, bias_eit, e_it_m, NE * SIT, SIT);
    moe_gemm_kernel<<<pgrid(NM * SIT), blk, 0, stream>>>(m_it_b, WeT, mask_mit, bias_mit, m_it_m, NM * SIT, SIT);

    // ---- Stage 4: cross-attn + gate + LayerNorm ----
    cross_gate_ln_kernel<<<dim3(NE), blk, 0, stream>>>(e_ti_m, etc_ori, Wg, bg, lns, lnb, ctx_e_ti, STI, 1);
    cross_gate_ln_kernel<<<dim3(NM), blk, 0, stream>>>(m_ti_m, mtc_ori, Wg, bg, lns, lnb, ctx_m_ti, STI, 0);
    cross_gate_ln_kernel<<<dim3(NE), blk, 0, stream>>>(e_it_m, eic_ori, Wg, bg, lns, lnb, ctx_e_it, SIT, 1);
    cross_gate_ln_kernel<<<dim3(NM), blk, 0, stream>>>(m_it_m, mic_ori, Wg, bg, lns, lnb, ctx_m_it, SIT, 0);

    // ---- Stage 5: final similarity scores ----
    score_kernel<<<dim3(64), blk, 0, stream>>>(ctx_m_ti, ctx_e_ti, ctx_m_it, ctx_e_it, out);
}
